// HAGMoE_68427418960311
// MI455X (gfx1250) — compile-verified, matrix operand order checked by reading, not run on hardware
//
#include <hip/hip_runtime.h>

#define USE_ASYNC 1   // global_load_async_to_lds_b128 via inline asm (assembler-verified in round 2)

// ---------------------------------------------------------------------------
// Types for CDNA5 WMMA
// ---------------------------------------------------------------------------
typedef __attribute__((ext_vector_type(16))) __bf16 v16bf;
typedef __attribute__((ext_vector_type(2)))  __bf16 v2bf;
typedef __attribute__((ext_vector_type(8)))  float  v8f;

union Frag {
    v16bf v;
    uint4 q[2];
};

// fp32 -> bf16 round-to-nearest-even (bit-trick fallback)
__device__ __forceinline__ unsigned bfbits(float f) {
    union { float f; unsigned u; } v; v.f = f;
    return (v.u + 0x7fffu + ((v.u >> 16) & 1u)) >> 16;
}

#if defined(__gfx1250__) && __has_builtin(__builtin_amdgcn_cvt_pk_bf16_f32)
#define HAVE_CVT_PK_BF16 1
#else
#define HAVE_CVT_PK_BF16 0
#endif

__device__ __forceinline__ unsigned pack2bf(float a, float b) {
#if HAVE_CVT_PK_BF16
    union { v2bf v; unsigned u; } r;
    r.v = __builtin_amdgcn_cvt_pk_bf16_f32(a, b);   // single v_cvt_pk_bf16_f32
    return r.u;
#else
    return bfbits(a) | (bfbits(b) << 16);
#endif
}
__device__ __forceinline__ __bf16 f2bf(float f) {
#if HAVE_CVT_PK_BF16
    v2bf r = __builtin_amdgcn_cvt_pk_bf16_f32(f, 0.f);
    return r[0];
#else
    union { unsigned short s; __bf16 b; } o;
    o.s = (unsigned short)bfbits(f);
    return o.b;
#endif
}

// Direct HBM -> LDS async copy (16B per lane), tracked by ASYNCcnt.
__device__ __forceinline__ void async_copy_b128(unsigned lds_off, const void* gptr) {
#if USE_ASYNC
    unsigned long long ga = (unsigned long long)(size_t)gptr;
    asm volatile("global_load_async_to_lds_b128 %0, %1, off"
                 :: "v"(lds_off), "v"(ga) : "memory");
#else
    (void)lds_off; (void)gptr;
#endif
}
__device__ __forceinline__ void async_wait0() {
#if USE_ASYNC
    asm volatile("s_wait_asynccnt 0x0" ::: "memory");
#endif
}

// ---------------------------------------------------------------------------
// Generic bf16 WMMA GEMM: C[M,N] = A[M,K](bf16) @ B[K,N](fp32->bf16)
//   BM=128, BN=128, BK=32, 256 threads = 8 waves, wave tile 64x32 (4x2 WMMA)
//   LDS double-buffered; A tile via async HBM->LDS DMA, B converted in regs.
// Options:
//   transB  : B[k,n] = Braw[n*ldb + k]              (for  v = a @ Wq^T)
//   expertN : B[k,nG] = Braw[(nG/expertN)*K*ldb + k*ldb + nG%expertN] (W1)
//   bias    : x += bias[n_global]   (b1 flat index == global column)
//   gelu    : exact-erf GELU
//   scale   : x *= scale[row*24 + n_global/scaleDiv]   (p_group*p_expert)
//   z-grid  : split-K / chunk offsets (ffn2)
// ---------------------------------------------------------------------------
__global__ __launch_bounds__(256)
void k_wmma_gemm(const __bf16* __restrict__ A, long lda, long aOffZ,
                 const float*  __restrict__ B, long ldb, long bOffZ,
                 int transB, int expertN,
                 const float*  __restrict__ bias,
                 const float*  __restrict__ scale, int scaleDiv,
                 int gelu,
                 float*  __restrict__ Cf, long ldcf, long cOffZ,
                 __bf16* __restrict__ Cbf, long ldcbf,
                 int K)
{
    constexpr int LSTR = 40;                 // padded bf16 stride (80B rows, 16B aligned)
    __shared__ __bf16 lsA[2][128 * LSTR];
    __shared__ __bf16 lsB[2][128 * LSTR];

    const int  tid = threadIdx.x;
    const long n0  = (long)blockIdx.x * 128;
    const long m0  = (long)blockIdx.y * 128;
    const long z   = (long)blockIdx.z;

    const __bf16* Ap = A + z * aOffZ + m0 * lda;
    const float*  Bz = B + z * bOffZ;
    const float*  Bp;
    if (expertN) {
        long ex = n0 / expertN;
        Bp = Bz + ex * (long)K * ldb + (n0 % expertN);
    } else if (transB) {
        Bp = Bz + n0 * ldb;
    } else {
        Bp = Bz + n0;
    }
    if (Cf) Cf += z * cOffZ;

    const int wave = tid >> 5, lane = tid & 31;
    const int wm  = (wave & 1) * 64;
    const int wn  = (wave >> 1) * 32;
    const int lm  = lane & 15;
    const int lk8 = (lane >> 4) << 3;        // 0 or 8

    const v8f vzero = {0.f, 0.f, 0.f, 0.f, 0.f, 0.f, 0.f, 0.f};
    v8f acc[4][2];
#pragma unroll
    for (int i = 0; i < 4; ++i)
#pragma unroll
        for (int j = 0; j < 2; ++j) acc[i][j] = vzero;

    // staging thread mappings
    const int arow = tid >> 1, acol = (tid & 1) << 4;       // A: 128 rows x 32 k (32B/thread)
    const int tbn  = tid >> 1, tbk  = (tid & 1) << 4;       // transB: 128 n x 32 k
    const int nbk2 = (tid >> 4) << 1;                       // normal: k pair 0,2,..,30
    const int nbn8 = (tid & 15) << 3;                       // normal: n octet 0..120

    // ---- A tile: async DMA straight into LDS (already bf16) ----
    auto stageA = [&](int buf, long k0) {
#if USE_ASYNC
        unsigned lds0 = (unsigned)(size_t)&lsA[buf][arow * LSTR + acol];
        const char* g = (const char*)(Ap + (long)arow * lda + k0 + acol);
        async_copy_b128(lds0,      g);
        async_copy_b128(lds0 + 16, g + 16);
#else
        const uint4* src = (const uint4*)(Ap + (long)arow * lda + k0 + acol);
        uint4 q0 = src[0], q1 = src[1];
        uint4* d = (uint4*)&lsA[buf][arow * LSTR + acol];
        d[0] = q0; d[1] = q1;
#endif
    };

    // ---- B tile: global -> regs (fp32), converted + stored after WMMA ----
    float brg[16];
    auto loadB = [&](long k0) {
        if (transB) {
            const float4* s = (const float4*)(Bp + (long)tbn * ldb + k0 + tbk);
#pragma unroll
            for (int i = 0; i < 4; ++i) {
                float4 f = s[i];
                brg[i * 4 + 0] = f.x; brg[i * 4 + 1] = f.y;
                brg[i * 4 + 2] = f.z; brg[i * 4 + 3] = f.w;
            }
        } else {
            const float4* s0 = (const float4*)(Bp + (k0 + nbk2) * ldb + nbn8);
            const float4* s1 = (const float4*)(Bp + (k0 + nbk2 + 1) * ldb + nbn8);
            float4 a0 = s0[0], a1 = s0[1], b0 = s1[0], b1 = s1[1];
            brg[0] = a0.x; brg[1] = a0.y; brg[2]  = a0.z; brg[3]  = a0.w;
            brg[4] = a1.x; brg[5] = a1.y; brg[6]  = a1.z; brg[7]  = a1.w;
            brg[8] = b0.x; brg[9] = b0.y; brg[10] = b0.z; brg[11] = b0.w;
            brg[12] = b1.x; brg[13] = b1.y; brg[14] = b1.z; brg[15] = b1.w;
        }
    };
    auto storeB = [&](int buf) {
        if (transB) {
            // 16 consecutive k at one n: pack pairs -> two uint4 stores
            unsigned u[8];
#pragma unroll
            for (int i = 0; i < 8; ++i) u[i] = pack2bf(brg[2 * i], brg[2 * i + 1]);
            uint4* d = (uint4*)&lsB[buf][tbn * LSTR + tbk];
            uint4 q0, q1;
            q0.x = u[0]; q0.y = u[1]; q0.z = u[2]; q0.w = u[3];
            q1.x = u[4]; q1.y = u[5]; q1.z = u[6]; q1.w = u[7];
            d[0] = q0; d[1] = q1;
        } else {
            // k-adjacent pair packed into one b32 per n
#pragma unroll
            for (int c = 0; c < 8; ++c)
                *(unsigned*)&lsB[buf][(nbn8 + c) * LSTR + nbk2] =
                    pack2bf(brg[c], brg[8 + c]);
        }
    };

    // ---- prologue: stage tile 0 ----
    stageA(0, 0);
    loadB(0);
    storeB(0);
    async_wait0();
    __syncthreads();

    int buf = 0;
    for (long k0 = 0; k0 < K; k0 += 32) {
        const bool more = (k0 + 32) < K;
        if (more) {
            stageA(buf ^ 1, k0 + 32);   // async DMA into other buffer
            loadB(k0 + 32);             // fp32 B into regs (latency hidden by WMMA)
        }

        Frag afr[4], bfr[2];
#pragma unroll
        for (int i = 0; i < 4; ++i) {
            const __bf16* r = &lsA[buf][(wm + i * 16 + lm) * LSTR];
            afr[i].q[0] = *(const uint4*)(r + lk8);
            afr[i].q[1] = *(const uint4*)(r + 16 + lk8);
        }
#pragma unroll
        for (int j = 0; j < 2; ++j) {
            const __bf16* r = &lsB[buf][(wn + j * 16 + lm) * LSTR];
            bfr[j].q[0] = *(const uint4*)(r + lk8);
            bfr[j].q[1] = *(const uint4*)(r + 16 + lk8);
        }
#pragma unroll
        for (int i = 0; i < 4; ++i)
#pragma unroll
            for (int j = 0; j < 2; ++j)
                acc[i][j] = __builtin_amdgcn_wmma_f32_16x16x32_bf16(
                    false, afr[i].v, false, bfr[j].v,
                    (short)0, acc[i][j], false, false);

        if (more) storeB(buf ^ 1);
        async_wait0();
        __syncthreads();
        buf ^= 1;
    }

    // ---- epilogue ----
#pragma unroll
    for (int i = 0; i < 4; ++i) {
#pragma unroll
        for (int j = 0; j < 2; ++j) {
#pragma unroll
            for (int r = 0; r < 8; ++r) {
                long mg = m0 + wm + i * 16 + ((lane >> 4) << 3) + r;
                long ng = n0 + wn + j * 16 + (lane & 15);
                float x = acc[i][j][r];
                if (bias)  x += bias[ng];
                if (gelu)  x = 0.5f * x * (1.0f + erff(x * 0.70710678118654752f));
                if (scale) x *= scale[mg * 24 + ng / scaleDiv];
                if (Cf)    Cf[mg * ldcf + ng] = x;
                if (Cbf)   Cbf[mg * ldcbf + ng] = f2bf(x);
            }
        }
    }
}

// ---------------------------------------------------------------------------
// aspect -> bf16 (GEMM A operand) and into cat2[:, 768:1536]
// ---------------------------------------------------------------------------
__global__ __launch_bounds__(256)
void k_cast_aspect(const float* __restrict__ a, __bf16* __restrict__ abf,
                   __bf16* __restrict__ cat2)
{
    int i = blockIdx.x * 256 + threadIdx.x;   // < 256*768
    int b = i / 768, h = i % 768;
    __bf16 v = f2bf(a[i]);
    abf[i] = v;
    cat2[(size_t)b * 1536 + 768 + h] = v;
}

// ---------------------------------------------------------------------------
// Per-batch: scores = (X[b]·v[b])/sqrt(H), mask, softmax over S, pool,
// then LayerNorm over cat(h_opinion, aspect) -> catn bf16 [B,1536]
// ---------------------------------------------------------------------------
__global__ __launch_bounds__(256)
void k_opinion(const float* __restrict__ X, const int* __restrict__ mask,
               const float* __restrict__ aspect, const float* __restrict__ v,
               const float* __restrict__ ln_g, const float* __restrict__ ln_b,
               __bf16* __restrict__ catn)
{
    __shared__ float sv[768], sop[768], ssc[128], red[256];
    const int b = blockIdx.x, tid = threadIdx.x;
    const float* Xb = X + (size_t)b * 128 * 768;

    for (int i = tid; i < 768; i += 256) sv[i] = v[b * 768 + i];
    __syncthreads();

    const int wave = tid >> 5, lane = tid & 31;
    for (int s = wave; s < 128; s += 8) {
        const float* xs = Xb + s * 768;
        float p = 0.f;
        for (int j = lane; j < 768; j += 32) p += xs[j] * sv[j];
#pragma unroll
        for (int off = 16; off; off >>= 1) p += __shfl_xor(p, off, 32);
        if (lane == 0) {
            float sc = p * 0.03608439182435161f;      // 1/sqrt(768)
            if (mask[b * 128 + s] == 0) sc = -3.0e38f;
            ssc[s] = sc;
        }
    }
    __syncthreads();

    if (wave == 0) {                                  // softmax over 128 by one wave
        float m4[4], mx = -3.4e38f;
#pragma unroll
        for (int i = 0; i < 4; ++i) { m4[i] = ssc[lane + 32 * i]; mx = fmaxf(mx, m4[i]); }
#pragma unroll
        for (int off = 16; off; off >>= 1) mx = fmaxf(mx, __shfl_xor(mx, off, 32));
        float e4[4], sum = 0.f;
#pragma unroll
        for (int i = 0; i < 4; ++i) { e4[i] = __expf(m4[i] - mx); sum += e4[i]; }
#pragma unroll
        for (int off = 16; off; off >>= 1) sum += __shfl_xor(sum, off, 32);
        float inv = 1.0f / sum;
#pragma unroll
        for (int i = 0; i < 4; ++i) ssc[lane + 32 * i] = e4[i] * inv;
    }
    __syncthreads();

    for (int h = tid; h < 768; h += 256) {            // pooled h_opinion
        float acc = 0.f;
        for (int s = 0; s < 128; ++s) acc += ssc[s] * Xb[s * 768 + h];
        sop[h] = acc;
    }
    __syncthreads();

    float s1 = 0.f, s2 = 0.f;                         // LayerNorm over 1536
    for (int i = tid; i < 1536; i += 256) {
        float x = (i < 768) ? sop[i] : aspect[b * 768 + i - 768];
        s1 += x; s2 += x * x;
    }
    red[tid] = s1; __syncthreads();
    for (int o = 128; o; o >>= 1) { if (tid < o) red[tid] += red[tid + o]; __syncthreads(); }
    const float mean = red[0] * (1.0f / 1536.0f);
    __syncthreads();
    red[tid] = s2; __syncthreads();
    for (int o = 128; o; o >>= 1) { if (tid < o) red[tid] += red[tid + o]; __syncthreads(); }
    const float var  = red[0] * (1.0f / 1536.0f) - mean * mean;
    const float rstd = rsqrtf(var + 1e-5f);

    for (int i = tid; i < 1536; i += 256) {
        float x = (i < 768) ? sop[i] : aspect[b * 768 + i - 768];
        catn[(size_t)b * 1536 + i] = f2bf((x - mean) * rstd * ln_g[i] + ln_b[i]);
    }
}

// ---------------------------------------------------------------------------
// Routing: w[b, g*8+e] = softmax_G(hf@Wgr+bgr)[g] * softmax_E((hc@Wer+ber)[g])[e]
// One wave per batch row.
// ---------------------------------------------------------------------------
__global__ __launch_bounds__(32)
void k_route(const float* __restrict__ hf, const float* __restrict__ hc,
             const float* __restrict__ Wgr, const float* __restrict__ bgr,
             const float* __restrict__ Wer, const float* __restrict__ ber,
             float* __restrict__ w)
{
    const int b = blockIdx.x, lane = threadIdx.x;
    float gl[3] = {0.f, 0.f, 0.f};
    float el[24];
#pragma unroll
    for (int i = 0; i < 24; ++i) el[i] = 0.f;

    for (int h = lane; h < 768; h += 32) {
        float f = hf[b * 768 + h], c = hc[b * 768 + h];
#pragma unroll
        for (int g = 0; g < 3; ++g) gl[g] += f * Wgr[h * 3 + g];
#pragma unroll
        for (int g = 0; g < 3; ++g)
#pragma unroll
            for (int e = 0; e < 8; ++e)
                el[g * 8 + e] += c * Wer[((size_t)g * 768 + h) * 8 + e];
    }
#pragma unroll
    for (int off = 16; off; off >>= 1) {
#pragma unroll
        for (int g = 0; g < 3; ++g) gl[g] += __shfl_xor(gl[g], off, 32);
#pragma unroll
        for (int i = 0; i < 24; ++i) el[i] += __shfl_xor(el[i], off, 32);
    }
    if (lane == 0) {
        float pg[3], mx = -3.4e38f, s = 0.f;
#pragma unroll
        for (int g = 0; g < 3; ++g) { gl[g] += bgr[g]; mx = fmaxf(mx, gl[g]); }
#pragma unroll
        for (int g = 0; g < 3; ++g) { pg[g] = __expf(gl[g] - mx); s += pg[g]; }
        float invs = 1.0f / s;
#pragma unroll
        for (int g = 0; g < 3; ++g) {
            float m2 = -3.4e38f, pe[8], s2 = 0.f;
#pragma unroll
            for (int e = 0; e < 8; ++e) { el[g*8+e] += ber[g*8+e]; m2 = fmaxf(m2, el[g*8+e]); }
#pragma unroll
            for (int e = 0; e < 8; ++e) { pe[e] = __expf(el[g*8+e] - m2); s2 += pe[e]; }
            float k = pg[g] * invs / s2;
#pragma unroll
            for (int e = 0; e < 8; ++e) w[b * 24 + g * 8 + e] = k * pe[e];
        }
    }
}

// ---------------------------------------------------------------------------
// out = h_fused + sum_z partial[z] + sum_ge w[b,ge]*b2[ge,h]
// ---------------------------------------------------------------------------
__global__ __launch_bounds__(256)
void k_merge(const float* __restrict__ hf, const float* __restrict__ part,
             const float* __restrict__ w, const float* __restrict__ b2,
             float* __restrict__ out)
{
    const int idx = blockIdx.x * 256 + threadIdx.x;   // < 196608
    const int b = idx / 768, h = idx % 768;
    float acc = hf[idx];
    for (int c = 0; c < 24; ++c) acc += part[(size_t)c * 196608 + idx];
    float bb = 0.f;
    for (int ge = 0; ge < 24; ++ge) bb += w[b * 24 + ge] * b2[ge * 768 + h];
    out[idx] = acc + bb;
}

// ---------------------------------------------------------------------------
extern "C" void kernel_launch(void* const* d_in, const int* in_sizes, int n_in,
                              void* d_out, int out_size, void* d_ws, size_t ws_size,
                              hipStream_t stream)
{
    (void)in_sizes; (void)n_in; (void)out_size; (void)ws_size;

    const float* X     = (const float*)d_in[0];
    const int*   mask  = (const int*)  d_in[1];
    const float* aspect= (const float*)d_in[2];
    const float* Wq    = (const float*)d_in[3];
    const float* ln_g  = (const float*)d_in[4];
    const float* ln_b  = (const float*)d_in[5];
    const float* Wf    = (const float*)d_in[6];
    const float* bfv   = (const float*)d_in[7];
    const float* Wgr   = (const float*)d_in[8];
    const float* bgr   = (const float*)d_in[9];
    const float* Wc    = (const float*)d_in[10];
    const float* bc    = (const float*)d_in[11];
    const float* Wer   = (const float*)d_in[12];
    const float* ber   = (const float*)d_in[13];
    const float* W1    = (const float*)d_in[14];
    const float* b1    = (const float*)d_in[15];
    const float* W2    = (const float*)d_in[16];
    const float* b2    = (const float*)d_in[17];
    float* out = (float*)d_out;

    char* ws = (char*)d_ws;
    size_t off = 0;
    auto alloc = [&](size_t bytes) -> void* {
        void* p = ws + off;
        off = (off + bytes + 255) & ~(size_t)255;
        return p;
    };
    __bf16* aspect_bf = (__bf16*)alloc((size_t)256 * 768 * 2);
    float*  vq        = (float*) alloc((size_t)256 * 768 * 4);
    __bf16* catn      = (__bf16*)alloc((size_t)256 * 1536 * 2);
    __bf16* cat2      = (__bf16*)alloc((size_t)256 * 1536 * 2);   // [hf_bf16 | aspect_bf16]
    float*  h_fused   = (float*) alloc((size_t)256 * 768 * 4);
    float*  h_cond    = (float*) alloc((size_t)256 * 768 * 4);
    float*  wgt       = (float*) alloc((size_t)256 * 24 * 4);
    __bf16* A2        = (__bf16*)alloc((size_t)256 * 73728 * 2);  // scaled gelu(mid)
    float*  part      = (float*) alloc((size_t)24 * 256 * 768 * 4);

    // 1) aspect -> bf16 (and cat2 right half)
    k_cast_aspect<<<768, 256, 0, stream>>>(aspect, aspect_bf, cat2);

    // 2) v = aspect @ Wq^T   [256,768] = [256,768]x[768,768]^T
    k_wmma_gemm<<<dim3(6, 2, 1), 256, 0, stream>>>(
        aspect_bf, 768, 0, Wq, 768, 0, /*transB=*/1, /*expertN=*/0,
        nullptr, nullptr, 1, /*gelu=*/0, vq, 768, 0, nullptr, 0, /*K=*/768);

    // 3) attention pooling + LayerNorm -> catn bf16
    k_opinion<<<256, 256, 0, stream>>>(X, mask, aspect, vq, ln_g, ln_b, catn);

    // 4) h_fused = catn @ Wf + bf  (fp32 + bf16 into cat2[:, :768])
    k_wmma_gemm<<<dim3(6, 2, 1), 256, 0, stream>>>(
        catn, 1536, 0, Wf, 768, 0, 0, 0,
        bfv, nullptr, 1, 0, h_fused, 768, 0, cat2, 1536, /*K=*/1536);

    // 5) h_cond = cat2 @ Wc + bc
    k_wmma_gemm<<<dim3(6, 2, 1), 256, 0, stream>>>(
        cat2, 1536, 0, Wc, 768, 0, 0, 0,
        bc, nullptr, 1, 0, h_cond, 768, 0, nullptr, 0, /*K=*/1536);

    // 6) routing weights w[b, g*8+e]
    k_route<<<256, 32, 0, stream>>>(h_fused, h_cond, Wgr, bgr, Wer, ber, wgt);

    // 7) FFN1: A2[b, n] = w[b, n/3072] * gelu(hf_bf16 @ W1 + b1), N = 24*3072
    k_wmma_gemm<<<dim3(576, 2, 1), 256, 0, stream>>>(
        cat2, 1536, 0, W1, 3072, 0, /*transB=*/0, /*expertN=*/3072,
        b1, wgt, 3072, /*gelu=*/1, nullptr, 0, 0, A2, 73728, /*K=*/768);

    // 8) FFN2 split-K over 24 expert chunks: part[z] = A2[:, z*3072:+3072] @ W2[z]
    k_wmma_gemm<<<dim3(6, 2, 24), 256, 0, stream>>>(
        A2, 73728, /*aOffZ=*/3072, W2, 768, /*bOffZ=*/(long)3072 * 768, 0, 0,
        nullptr, nullptr, 1, 0, part, 768, /*cOffZ=*/(long)256 * 768, nullptr, 0,
        /*K=*/3072);

    // 9) merge: residual + partial sum + weighted b2
    k_merge<<<768, 256, 0, stream>>>(h_fused, part, wgt, b2, out);
}
